// ViLLayer_61744449847850
// MI455X (gfx1250) — compile-verified, matrix operand order checked by reading, not run on hardware
//
#include <hip/hip_runtime.h>

// ---------------------------------------------------------------------------
// Types for CDNA5 WMMA (gfx1250, wave32)
// ---------------------------------------------------------------------------
typedef __attribute__((ext_vector_type(16))) __bf16 v16bf;
typedef __attribute__((ext_vector_type(8)))  float  v8f;

union FragBF { v16bf v; unsigned short u[16]; };
union AccF   { v8f   v; float          f[8];  };

constexpr int B_     = 4;
constexpr int S_     = 1024;
constexpr int DIM_   = 512;
constexpr int INNER_ = 1024;
constexpr int NH_    = 16;
constexpr int DH_    = 64;
constexpr int KCONV  = 4;
constexpr int M_     = B_ * S_;        // 4096 token rows
constexpr int UP_N   = 2 * INNER_;     // 2048

__device__ __forceinline__ unsigned short f2bf(float f) {
  union { float f; unsigned u; } x; x.f = f;
  unsigned r = x.u + 0x7FFFu + ((x.u >> 16) & 1u);   // round-to-nearest-even
  return (unsigned short)(r >> 16);
}
__device__ __forceinline__ float bf2f(unsigned short s) {
  union { float f; unsigned u; } x; x.u = ((unsigned)s) << 16; return x.f;
}
__device__ __forceinline__ v8f wmma_bf16(v16bf a, v16bf b, v8f c) {
  return __builtin_amdgcn_wmma_f32_16x16x32_bf16(false, a, false, b, (short)0, c,
                                                 false, false);
}
// CDNA5 async global->LDS copy (ASYNCcnt-tracked, no VGPR data)
__device__ __forceinline__ void async_load_b128(unsigned lds_off, const void* gptr) {
  asm volatile("global_load_async_to_lds_b128 %0, %1, off"
               :: "v"(lds_off), "v"(gptr) : "memory");
}
__device__ __forceinline__ void wait_async0() {
  asm volatile("s_wait_asynccnt 0x0" ::: "memory");
}

// ---------------------------------------------------------------------------
// 0) f32 -> bf16 conversion
// ---------------------------------------------------------------------------
__global__ void cvt_bf16_kernel(const float* __restrict__ in,
                                unsigned short* __restrict__ out, int n) {
  int i = blockIdx.x * blockDim.x + threadIdx.x;
  if (i < n) out[i] = f2bf(in[i]);
}

// ---------------------------------------------------------------------------
// 1) bf16 WMMA GEMM:  C[M,N] = A[M,K] * Bm[N,K]^T + bias
//    128-thread block = 4 waves computes a 64x64 tile. The 64-row x 32-col
//    B K-slice is staged into LDS with global_load_async_to_lds_b128 and
//    shared by all 4 waves; each wave reuses its A fragment across 4 wmmas.
// ---------------------------------------------------------------------------
constexpr int LDSB_STRIDE = 80;  // 64 data bytes + 16 pad (bank spread)

__global__ void gemm_bf16_kernel(const unsigned short* __restrict__ A,
                                 const unsigned short* __restrict__ Bm,
                                 const float* __restrict__ bias,
                                 float* __restrict__ C, int Kd, int N) {
  __shared__ __align__(16) unsigned char ldsB[64 * LDSB_STRIDE];
  const int tid  = threadIdx.x;            // 0..127
  const int wave = tid >> 5;               // 0..3 -> M sub-tile
  const int lane = tid & 31;
  const int rc = lane & 15;
  const int hi = lane >> 4;
  const int k0 = hi ? 8 : 0;
  const int n0 = blockIdx.x * 64;
  const int m0 = blockIdx.y * 64 + wave * 16;
  const unsigned char* Bbytes = (const unsigned char*)Bm;
  const unsigned ldsbase = (unsigned)(uintptr_t)&ldsB[0];
  const unsigned short* arow = A + (size_t)(m0 + rc) * Kd;

  AccF acc[4];
#pragma unroll
  for (int n = 0; n < 4; ++n)
#pragma unroll
    for (int r = 0; r < 8; ++r) acc[n].f[r] = 0.f;

  for (int kk = 0; kk < Kd; kk += 32) {
    __syncthreads();   // previous iteration's LDS reads complete
    // stage B[n0..n0+63, kk..kk+31] into LDS: 256 16-byte chunks / 128 threads
#pragma unroll
    for (int c = 0; c < 2; ++c) {
      int chunk = tid + c * 128;
      int row = chunk >> 2, part = chunk & 3;
      async_load_b128(ldsbase + row * LDSB_STRIDE + part * 16,
                      Bbytes + (((size_t)(n0 + row) * Kd + kk) * 2 + part * 16));
    }
    wait_async0();
    __syncthreads();
    if (kk + 32 < Kd)  // speculative prefetch of next A K-block
      __builtin_prefetch(arow + kk + 32, 0, 0);
    FragBF a;
#pragma unroll
    for (int i = 0; i < 8; ++i) {
      a.u[i]     = arow[kk + k0 + i];
      a.u[8 + i] = arow[kk + k0 + 16 + i];
    }
#pragma unroll
    for (int n = 0; n < 4; ++n) {
      const unsigned short* brow =
          (const unsigned short*)&ldsB[(n * 16 + rc) * LDSB_STRIDE];
      FragBF b;
#pragma unroll
      for (int i = 0; i < 8; ++i) {
        b.u[i]     = brow[k0 + i];
        b.u[8 + i] = brow[k0 + 16 + i];
      }
      acc[n].v = wmma_bf16(a.v, b.v, acc[n].v);
    }
  }
#pragma unroll
  for (int n = 0; n < 4; ++n) {
    const float bv = bias ? bias[n0 + n * 16 + rc] : 0.f;
#pragma unroll
    for (int r = 0; r < 8; ++r)
      C[(size_t)(m0 + r + 8 * hi) * N + n0 + n * 16 + rc] = acc[n].f[r] + bv;
  }
}

// ---------------------------------------------------------------------------
// 2) Depthwise causal conv + SiLU; also emit bf16 copies of xa and x_m
// ---------------------------------------------------------------------------
__global__ void conv_silu_kernel(const float* __restrict__ xz,
                                 const float* __restrict__ cw,
                                 const float* __restrict__ cb,
                                 float* __restrict__ xa,
                                 unsigned short* __restrict__ xabf,
                                 unsigned short* __restrict__ xmbf) {
  int idx = blockIdx.x * blockDim.x + threadIdx.x;   // B*S*INNER
  int c = idx % INNER_;
  int s = (idx / INNER_) % S_;
  int b = idx / (INNER_ * S_);
  size_t row = (size_t)b * S_ + s;
  float acc = cb[c];
#pragma unroll
  for (int j = 0; j < KCONV; ++j) {
    int sp = s - (KCONV - 1) + j;
    if (sp >= 0) acc += xz[((size_t)b * S_ + sp) * UP_N + c] * cw[c * KCONV + j];
  }
  float sl = acc / (1.f + __expf(-acc));
  xa  [row * INNER_ + c] = sl;
  xabf[row * INNER_ + c] = f2bf(sl);
  xmbf[row * INNER_ + c] = f2bf(xz[row * UP_N + c]);
}

// ---------------------------------------------------------------------------
// 3) Headwise 64x64 projection per head (WMMA). outT!=null => also write
//    transposed (B,NH,DH,S) layout (for the C@V operand of mLSTM).
// ---------------------------------------------------------------------------
__global__ void headwise_kernel(const unsigned short* __restrict__ in,
                                const unsigned short* __restrict__ W,
                                unsigned short* __restrict__ out,
                                unsigned short* __restrict__ outT) {
  const int lane = threadIdx.x & 31;
  const int rc = lane & 15, hi = lane >> 4, k0 = hi ? 8 : 0;
  const int m0 = blockIdx.x * 16;
  const int n0 = blockIdx.y * 16;
  const int hh = blockIdx.z;
  const unsigned short* arow = in + (size_t)(m0 + rc) * INNER_ + hh * DH_;
  const unsigned short* brow = W + ((size_t)hh * DH_ + n0 + rc) * DH_;
  AccF acc;
#pragma unroll
  for (int r = 0; r < 8; ++r) acc.f[r] = 0.f;
#pragma unroll
  for (int kk = 0; kk < 2; ++kk) {
    FragBF a, b;
#pragma unroll
    for (int i = 0; i < 8; ++i) {
      a.u[i]     = arow[kk * 32 + k0 + i];
      a.u[8 + i] = arow[kk * 32 + k0 + 16 + i];
      b.u[i]     = brow[kk * 32 + k0 + i];
      b.u[8 + i] = brow[kk * 32 + k0 + 16 + i];
    }
    acc.v = wmma_bf16(a.v, b.v, acc.v);
  }
#pragma unroll
  for (int r = 0; r < 8; ++r) {
    int m = m0 + r + 8 * hi;
    unsigned short val = f2bf(acc.f[r]);
    out[(size_t)m * INNER_ + hh * DH_ + n0 + rc] = val;
    if (outT) {
      int b = m / S_, s = m % S_;
      outT[(((size_t)(b * NH_ + hh)) * DH_ + n0 + rc) * S_ + s] = val;
    }
  }
}

// ---------------------------------------------------------------------------
// 4) Gate projections (wave per (b,h,s)): ip and log-sigmoid(fp)
// ---------------------------------------------------------------------------
__global__ void gate_kernel(const unsigned short* __restrict__ q,
                            const unsigned short* __restrict__ k,
                            const unsigned short* __restrict__ v,
                            const float* __restrict__ Wi, const float* __restrict__ bi,
                            const float* __restrict__ Wf, const float* __restrict__ bf_,
                            float* __restrict__ ip, float* __restrict__ logf) {
  int gw = (blockIdx.x * blockDim.x + threadIdx.x) >> 5;
  int lane = threadIdx.x & 31;
  int s = gw % S_;
  int hh = (gw / S_) % NH_;
  int b = gw / (S_ * NH_);
  size_t row = ((size_t)b * S_ + s) * INNER_;
  const float* wi = Wi + (size_t)hh * 3 * INNER_;
  const float* wf = Wf + (size_t)hh * 3 * INNER_;
  float ai = 0.f, af = 0.f;
  for (int j = lane; j < INNER_; j += 32) {
    float qv = bf2f(q[row + j]), kv = bf2f(k[row + j]), vv = bf2f(v[row + j]);
    ai += qv * wi[j] + kv * wi[INNER_ + j] + vv * wi[2 * INNER_ + j];
    af += qv * wf[j] + kv * wf[INNER_ + j] + vv * wf[2 * INNER_ + j];
  }
#pragma unroll
  for (int m = 16; m >= 1; m >>= 1) {
    ai += __shfl_xor(ai, m, 32);
    af += __shfl_xor(af, m, 32);
  }
  if (lane == 0) {
    size_t o = ((size_t)b * NH_ + hh) * S_ + s;
    ip[o] = ai + bi[hh];
    float fp = af + bf_[hh];
    logf[o] = (fp >= 0.f) ? -log1pf(__expf(-fp)) : (fp - log1pf(__expf(fp)));
  }
}

// ---------------------------------------------------------------------------
// 5) Inclusive prefix sum of logf over S, per (b,h). One block per (b,h).
// ---------------------------------------------------------------------------
__global__ void scan_kernel(const float* __restrict__ logf, float* __restrict__ cum) {
  __shared__ float tot[256];
  int bh = blockIdx.x, t = threadIdx.x;
  const float* src = logf + (size_t)bh * S_;
  float* dst = cum + (size_t)bh * S_;
  float v0 = src[t * 4 + 0];
  float v1 = v0 + src[t * 4 + 1];
  float v2 = v1 + src[t * 4 + 2];
  float v3 = v2 + src[t * 4 + 3];
  tot[t] = v3;
  __syncthreads();
  if (t == 0) {
    float a = 0.f;
    for (int i = 0; i < 256; ++i) { float x = tot[i]; tot[i] = a; a += x; }
  }
  __syncthreads();
  float o = tot[t];
  dst[t * 4 + 0] = v0 + o;
  dst[t * 4 + 1] = v1 + o;
  dst[t * 4 + 2] = v2 + o;
  dst[t * 4 + 3] = v3 + o;
}

// ---------------------------------------------------------------------------
// 6) Flash-style stabilized parallel mLSTM. Wave per 16-row q tile.
//    Two WMMA stages per 32-wide t block, C packed through LDS to bf16.
// ---------------------------------------------------------------------------
__global__ void mlstm_kernel(const unsigned short* __restrict__ q,
                             const unsigned short* __restrict__ k,
                             const unsigned short* __restrict__ vT,
                             const float* __restrict__ cum,
                             const float* __restrict__ ip,
                             float* __restrict__ h) {
  __shared__ unsigned short cls[16][34];
  const int st = blockIdx.x, hh = blockIdx.y, b = blockIdx.z;
  const int lane = threadIdx.x & 31;
  const int rc = lane & 15, hi = lane >> 4, k0 = hi ? 8 : 0;
  const int s0 = st * 16;
  const size_t bhS = ((size_t)b * NH_ + hh) * S_;

  FragBF qa[2];
  {
    const unsigned short* qrow = q + ((size_t)b * S_ + s0 + rc) * INNER_ + hh * DH_;
#pragma unroll
    for (int kk = 0; kk < 2; ++kk)
#pragma unroll
      for (int i = 0; i < 8; ++i) {
        qa[kk].u[i]     = qrow[kk * 32 + k0 + i];
        qa[kk].u[8 + i] = qrow[kk * 32 + k0 + 16 + i];
      }
  }
  float mrow[8], drow[8], cums[8];
  AccF accN[4];
#pragma unroll
  for (int r = 0; r < 8; ++r) {
    mrow[r] = -1e30f; drow[r] = 0.f;
    cums[r] = cum[bhS + s0 + r + 8 * hi];
  }
#pragma unroll
  for (int n = 0; n < 4; ++n)
#pragma unroll
    for (int r = 0; r < 8; ++r) accN[n].f[r] = 0.f;

  const float invsq = 0.125f;  // 1/sqrt(64)
  const int tmax = s0 + 15;
  for (int t0 = 0; t0 <= tmax; t0 += 32) {
    // ---- QK^T over d=64 (two 16x16 subtiles in t) ----
    AccF qk[2];
#pragma unroll
    for (int sub = 0; sub < 2; ++sub) {
#pragma unroll
      for (int r = 0; r < 8; ++r) qk[sub].f[r] = 0.f;
      const unsigned short* krow =
          k + ((size_t)b * S_ + t0 + sub * 16 + rc) * INNER_ + hh * DH_;
#pragma unroll
      for (int kk = 0; kk < 2; ++kk) {
        FragBF kb;
#pragma unroll
        for (int i = 0; i < 8; ++i) {
          kb.u[i]     = krow[kk * 32 + k0 + i];
          kb.u[8 + i] = krow[kk * 32 + k0 + 16 + i];
        }
        qk[sub].v = wmma_bf16(qa[kk].v, kb.v, qk[sub].v);
      }
    }
    // ---- gating + online max stabilization ----
    float cumt[2], ipt[2];
#pragma unroll
    for (int sub = 0; sub < 2; ++sub) {
      cumt[sub] = cum[bhS + t0 + sub * 16 + rc];
      ipt[sub]  = ip [bhS + t0 + sub * 16 + rc];
    }
    float logd[2][8], tm[8];
#pragma unroll
    for (int r = 0; r < 8; ++r) tm[r] = -1e30f;
#pragma unroll
    for (int sub = 0; sub < 2; ++sub)
#pragma unroll
      for (int r = 0; r < 8; ++r) {
        int srow = s0 + r + 8 * hi;
        int tcol = t0 + sub * 16 + rc;
        float ld = (tcol <= srow) ? (cums[r] - cumt[sub] + ipt[sub]) : -1e30f;
        logd[sub][r] = ld;
        tm[r] = fmaxf(tm[r], ld);
      }
#pragma unroll
    for (int r = 0; r < 8; ++r)
#pragma unroll
      for (int m = 8; m >= 1; m >>= 1)
        tm[r] = fmaxf(tm[r], __shfl_xor(tm[r], m, 32));
    float scl[8];
#pragma unroll
    for (int r = 0; r < 8; ++r) {
      float mn = fmaxf(mrow[r], tm[r]);
      scl[r] = __expf(mrow[r] - mn);
      mrow[r] = mn;
      drow[r] *= scl[r];
    }
#pragma unroll
    for (int n = 0; n < 4; ++n)
#pragma unroll
      for (int r = 0; r < 8; ++r) accN[n].f[r] *= scl[r];
    // ---- C = qk * exp(logD - m); row-sum + pack to bf16 A layout via LDS ----
    float rs[8];
#pragma unroll
    for (int r = 0; r < 8; ++r) rs[r] = 0.f;
#pragma unroll
    for (int sub = 0; sub < 2; ++sub)
#pragma unroll
      for (int r = 0; r < 8; ++r) {
        float cel = qk[sub].f[r] * invsq * __expf(logd[sub][r] - mrow[r]);
        rs[r] += cel;
        cls[r + 8 * hi][sub * 16 + rc] = f2bf(cel);
      }
#pragma unroll
    for (int r = 0; r < 8; ++r)
#pragma unroll
      for (int m = 8; m >= 1; m >>= 1)
        rs[r] += __shfl_xor(rs[r], m, 32);
#pragma unroll
    for (int r = 0; r < 8; ++r) drow[r] += rs[r];
    __syncthreads();
    FragBF ca;
#pragma unroll
    for (int i = 0; i < 8; ++i) {
      ca.u[i]     = cls[rc][k0 + i];
      ca.u[8 + i] = cls[rc][k0 + 16 + i];
    }
    __syncthreads();
    // ---- numerator += C(16x32) @ V(32x64) ----
#pragma unroll
    for (int n = 0; n < 4; ++n) {
      const unsigned short* vrow =
          vT + (((size_t)b * NH_ + hh) * DH_ + n * 16 + rc) * S_ + t0;
      FragBF vb;
#pragma unroll
      for (int i = 0; i < 8; ++i) {
        vb.u[i]     = vrow[k0 + i];
        vb.u[8 + i] = vrow[k0 + 16 + i];
      }
      accN[n].v = wmma_bf16(ca.v, vb.v, accN[n].v);
    }
  }
  // ---- final normalizer and store h (B,NH,S,DH) ----
#pragma unroll
  for (int r = 0; r < 8; ++r) {
    float nr = fmaxf(fabsf(drow[r]), __expf(-mrow[r])) + 5e-5f;
    float inv = 1.f / nr;
    int srow = s0 + r + 8 * hi;
#pragma unroll
    for (int n = 0; n < 4; ++n)
      h[(bhS + srow) * DH_ + n * 16 + rc] = accN[n].f[r] * inv;
  }
}

// ---------------------------------------------------------------------------
// 7) Multi-head LayerNorm + skip + SiLU(z)  -> bf16 rows for down-proj
// ---------------------------------------------------------------------------
__global__ void outnorm_kernel(const float* __restrict__ h,
                               const float* __restrict__ ln_w,
                               const float* __restrict__ ln_b,
                               const float* __restrict__ skip,
                               const float* __restrict__ xa,
                               const float* __restrict__ xz,
                               unsigned short* __restrict__ hsbf) {
  int idx = blockIdx.x * blockDim.x + threadIdx.x;   // B*S*NH
  int hh = idx % NH_;
  int s = (idx / NH_) % S_;
  int b = idx / (NH_ * S_);
  const float* hr = h + (((size_t)b * NH_ + hh) * S_ + s) * DH_;
  float mu = 0.f;
  for (int d = 0; d < DH_; ++d) mu += hr[d];
  mu *= (1.f / DH_);
  float var = 0.f;
  for (int d = 0; d < DH_; ++d) { float t = hr[d] - mu; var += t * t; }
  var *= (1.f / DH_);
  float rstd = rsqrtf(var + 1e-3f);
  size_t row = (size_t)b * S_ + s;
  for (int d = 0; d < DH_; ++d) {
    int c = hh * DH_ + d;
    float hn = (hr[d] - mu) * rstd * (1.f + ln_w[c]) + ln_b[c];
    float z = xz[row * UP_N + INNER_ + c];
    float sz = z / (1.f + __expf(-z));
    float out = (hn + skip[c] * xa[row * INNER_ + c]) * sz;
    hsbf[row * INNER_ + c] = f2bf(out);
  }
}

// ---------------------------------------------------------------------------
// Host: orchestrate the pipeline on `stream`
// ---------------------------------------------------------------------------
extern "C" void kernel_launch(void* const* d_in, const int* in_sizes, int n_in,
                              void* d_out, int out_size, void* d_ws, size_t ws_size,
                              hipStream_t stream) {
  (void)in_sizes; (void)n_in; (void)out_size; (void)ws_size;
  const float* x      = (const float*)d_in[0];
  const float* W_up   = (const float*)d_in[1];
  const float* b_up   = (const float*)d_in[2];
  const float* W_q    = (const float*)d_in[3];
  const float* W_k    = (const float*)d_in[4];
  const float* W_v    = (const float*)d_in[5];
  const float* conv_w = (const float*)d_in[6];
  const float* conv_b = (const float*)d_in[7];
  const float* W_i    = (const float*)d_in[8];
  const float* b_i    = (const float*)d_in[9];
  const float* W_f    = (const float*)d_in[10];
  const float* b_f    = (const float*)d_in[11];
  const float* ln_w   = (const float*)d_in[12];
  const float* ln_b   = (const float*)d_in[13];
  const float* skip   = (const float*)d_in[14];
  const float* W_down = (const float*)d_in[15];
  const float* b_down = (const float*)d_in[16];

  char* ws = (char*)d_ws;
  size_t off = 0;
  auto alloc = [&](size_t bytes) -> char* {
    char* p = ws + off;
    off += (bytes + 255) & ~(size_t)255;
    return p;
  };
  unsigned short* xbf   = (unsigned short*)alloc((size_t)M_ * DIM_ * 2);
  unsigned short* wupbf = (unsigned short*)alloc((size_t)UP_N * DIM_ * 2);
  unsigned short* wqbf  = (unsigned short*)alloc((size_t)NH_ * DH_ * DH_ * 2);
  unsigned short* wkbf  = (unsigned short*)alloc((size_t)NH_ * DH_ * DH_ * 2);
  unsigned short* wvbf  = (unsigned short*)alloc((size_t)NH_ * DH_ * DH_ * 2);
  unsigned short* wdbf  = (unsigned short*)alloc((size_t)DIM_ * INNER_ * 2);
  float*          xz    = (float*)alloc((size_t)M_ * UP_N * 4);
  float*          xa    = (float*)alloc((size_t)M_ * INNER_ * 4);
  unsigned short* xabf  = (unsigned short*)alloc((size_t)M_ * INNER_ * 2);
  unsigned short* xmbf  = (unsigned short*)alloc((size_t)M_ * INNER_ * 2);
  unsigned short* qbf   = (unsigned short*)alloc((size_t)M_ * INNER_ * 2);
  unsigned short* kbf   = (unsigned short*)alloc((size_t)M_ * INNER_ * 2);
  unsigned short* vbf   = (unsigned short*)alloc((size_t)M_ * INNER_ * 2);
  unsigned short* vTbf  = (unsigned short*)alloc((size_t)M_ * INNER_ * 2);
  float*          ipb   = (float*)alloc((size_t)B_ * NH_ * S_ * 4);
  float*          logf  = (float*)alloc((size_t)B_ * NH_ * S_ * 4);
  float*          cumb  = (float*)alloc((size_t)B_ * NH_ * S_ * 4);
  float*          hbuf  = (float*)alloc((size_t)B_ * NH_ * S_ * DH_ * 4);
  unsigned short* hsbf  = (unsigned short*)alloc((size_t)M_ * INNER_ * 2);

  auto cvt = [&](const float* src, unsigned short* dst, int n) {
    cvt_bf16_kernel<<<(n + 255) / 256, 256, 0, stream>>>(src, dst, n);
  };
  // 0) conversions
  cvt(x,      xbf,   M_ * DIM_);
  cvt(W_up,   wupbf, UP_N * DIM_);
  cvt(W_q,    wqbf,  NH_ * DH_ * DH_);
  cvt(W_k,    wkbf,  NH_ * DH_ * DH_);
  cvt(W_v,    wvbf,  NH_ * DH_ * DH_);
  cvt(W_down, wdbf,  DIM_ * INNER_);

  // 1) up projection: xz = x @ W_up^T + b_up   (64x64 tile per 128-thr block)
  gemm_bf16_kernel<<<dim3(UP_N / 64, M_ / 64), 128, 0, stream>>>(
      xbf, wupbf, b_up, xz, DIM_, UP_N);

  // 2) causal conv + silu
  conv_silu_kernel<<<(M_ * INNER_) / 256, 256, 0, stream>>>(
      xz, conv_w, conv_b, xa, xabf, xmbf);

  // 3) headwise q, k, v
  headwise_kernel<<<dim3(M_ / 16, DH_ / 16, NH_), 32, 0, stream>>>(
      xabf, wqbf, qbf, (unsigned short*)nullptr);
  headwise_kernel<<<dim3(M_ / 16, DH_ / 16, NH_), 32, 0, stream>>>(
      xabf, wkbf, kbf, (unsigned short*)nullptr);
  headwise_kernel<<<dim3(M_ / 16, DH_ / 16, NH_), 32, 0, stream>>>(
      xmbf, wvbf, vbf, vTbf);

  // 4) gate projections
  gate_kernel<<<(B_ * NH_ * S_) / 8, 256, 0, stream>>>(
      qbf, kbf, vbf, W_i, b_i, W_f, b_f, ipb, logf);

  // 5) cumulative sum of log-sigmoid(f)
  scan_kernel<<<B_ * NH_, 256, 0, stream>>>(logf, cumb);

  // 6) mLSTM
  mlstm_kernel<<<dim3(S_ / 16, NH_, B_), 32, 0, stream>>>(
      qbf, kbf, vTbf, cumb, ipb, hbuf);

  // 7) norm + skip + gate
  outnorm_kernel<<<(B_ * S_ * NH_) / 256, 256, 0, stream>>>(
      hbuf, ln_w, ln_b, skip, xa, xz, hsbf);

  // 8) down projection into d_out
  gemm_bf16_kernel<<<dim3(DIM_ / 64, M_ / 64), 128, 0, stream>>>(
      hsbf, wdbf, b_down, (float*)d_out, INNER_, DIM_);
}
